// QKVAttention_39599598469377
// MI455X (gfx1250) — matrix operand order repaired and checked by reading, not compiled, not run on hardware
//
#include <hip/hip_runtime.h>
#include <hip/hip_bf16.h>

typedef __attribute__((ext_vector_type(16))) __bf16 v16bf;
typedef __attribute__((ext_vector_type(8)))  __bf16 v8bf;
typedef __attribute__((ext_vector_type(4)))  __bf16 v4bf;
typedef __attribute__((ext_vector_type(8)))  float  v8f;

#define NW      4            // waves per block (wave32)
#define THREADS (NW * 32)
#define BLOCKS  250
#define ITERS   4            // BLOCKS*ITERS*NW = 4000 token pairs = 8000 tokens
#define Fn      128
#define An      128
#define Cn      8
#define XSTR    132          // f32 LDS row stride (padded, 16B aligned rows)
#define HSTR    136          // bf16 LDS row stride (padded, 16B aligned rows)

__device__ __forceinline__ v16bf ld_bf16x16(const __bf16* p) {
    // two aligned 16B LDS loads -> one 16-element bf16 fragment
    v8bf lo = *(const v8bf*)p;
    v8bf hi = *(const v8bf*)(p + 16);
    return __builtin_shufflevector(lo, hi, 0, 1, 2, 3, 4, 5, 6, 7,
                                           8, 9, 10, 11, 12, 13, 14, 15);
}

__global__ __launch_bounds__(THREADS) void qkv_chan_attn_kernel(
    const float* __restrict__ x,  const float* __restrict__ Wq, const float* __restrict__ bq,
    const float* __restrict__ Wk, const float* __restrict__ bk,
    const float* __restrict__ Wv, const float* __restrict__ bv,
    float* __restrict__ out)
{
    // weights resident in LDS as bf16, TRANSPOSED [n][k] so B-fragments are contiguous
    __shared__ __bf16 sWqT[An * HSTR];       // ~34 KB
    __shared__ __bf16 sWkT[An * HSTR];       // ~34 KB
    __shared__ float  sWv[Fn];
    __shared__ float  sBq[An];
    __shared__ float  sBk[An];
    __shared__ float  sBv;
    // per-wave working tiles
    __shared__ float  sX [NW][16 * XSTR];    // x tile f32 (v-dot + final sum)
    __shared__ __bf16 sXb[NW][16 * HSTR];    // x tile bf16 (A-fragments)
    __shared__ __bf16 sQ [NW][16 * HSTR];    // q tile, row-major bf16
    __shared__ __bf16 sK [NW][16 * HSTR];    // k tile, row-major bf16
    __shared__ float  sS [NW][16 * 16];      // score tile
    __shared__ float  sV [NW][16];           // v per row
    __shared__ float  sW [NW][16];           // pre-softmax channel weights

    const int tid  = threadIdx.x;
    const int lane = tid & 31;
    const int wv   = tid >> 5;

    // ---- cooperative weight preload: coalesced global read, transposed LDS write ----
    #pragma unroll 4
    for (int i = tid; i < Fn * An; i += THREADS) {
        int k = i >> 7, n = i & 127;
        sWqT[n * HSTR + k] = (__bf16)Wq[i];
        sWkT[n * HSTR + k] = (__bf16)Wk[i];
    }
    if (tid < Fn) { sWv[tid] = Wv[tid]; sBq[tid] = bq[tid]; sBk[tid] = bk[tid]; }
    if (tid == 0) sBv = bv[0];
    __syncthreads();

    float*  sXw  = sX [wv];
    __bf16* sXbw = sXb[wv];
    __bf16* sQw  = sQ [wv];
    __bf16* sKw  = sK [wv];
    float*  sSw  = sS [wv];
    float*  sVw  = sV [wv];
    float*  sWw  = sW [wv];

    // WMMA lane decomposition (wave32, 16x16x32 bf16)
    const int m   = lane & 15;          // A-matrix row
    const int khA = (lane >> 4) * 8;    // A-matrix K sub-base within 32-chunk
    const int nB  = lane & 15;          // B-matrix column
    const int kqB = (lane >> 4) * 8;    // B-matrix K sub-base within 32-chunk
    const int mD  = (lane >> 4) * 8;    // C/D row offset

    for (int it = 0; it < ITERS; ++it) {
        const int pair = blockIdx.x * (ITERS * NW) + it * NW + wv;

        // ---- coalesced x tile load: 512 float4, 16 per lane; dual f32 + bf16 store ----
        const float4* xg4 = (const float4*)(x + (size_t)pair * (2 * Cn * Fn));
        #pragma unroll
        for (int i = 0; i < 16; ++i) {
            int gi = lane + i * 32;
            float4 vx = xg4[gi];
            int row = gi >> 5, c4 = gi & 31;
            *(float4*)&sXw[row * XSTR + c4 * 4] = vx;
            v4bf hb;
            hb[0] = (__bf16)vx.x; hb[1] = (__bf16)vx.y;
            hb[2] = (__bf16)vx.z; hb[3] = (__bf16)vx.w;
            *(v4bf*)&sXbw[row * HSTR + c4 * 4] = hb;
        }
        __syncthreads();

        // ---- A-fragments of x: two b128 LDS loads each (reused for Q and K) ----
        v16bf a[4];
        #pragma unroll
        for (int kc = 0; kc < 4; ++kc) {
            const __bf16* pa = &sXbw[m * HSTR + kc * 32 + khA];
            v8bf lo = *(const v8bf*)pa;           // K = base .. base+7
            v8bf hi = *(const v8bf*)(pa + 16);    // K = base+16 .. base+23
            a[kc] = __builtin_shufflevector(lo, hi, 0, 1, 2, 3, 4, 5, 6, 7,
                                                    8, 9, 10, 11, 12, 13, 14, 15);
        }

        // ---- Q = x@Wq + bq, K = x@Wk + bk  (8 N-tiles x 4 K-chunks x 2) ----
        #pragma unroll
        for (int nt = 0; nt < 8; ++nt) {
            v8f accq = {};
            v8f acck = {};
            const int nrow = (nt * 16 + nB) * HSTR;
            #pragma unroll
            for (int kc = 0; kc < 4; ++kc) {
                int kk = kc * 32 + kqB;
                v16bf bqf = ld_bf16x16(&sWqT[nrow + kk]);
                v16bf bkf = ld_bf16x16(&sWkT[nrow + kk]);
                accq = __builtin_amdgcn_wmma_f32_16x16x32_bf16(
                           false, a[kc], false, bqf, (short)0, accq, false, false);
                acck = __builtin_amdgcn_wmma_f32_16x16x32_bf16(
                           false, a[kc], false, bkf, (short)0, acck, false, false);
            }
            #pragma unroll
            for (int r = 0; r < 8; ++r) {
                int M = r + mD, nc = nt * 16 + nB;
                sQw[M * HSTR + nc] = (__bf16)(accq[r] + sBq[nc]);
                sKw[M * HSTR + nc] = (__bf16)(acck[r] + sBk[nc]);
            }
        }
        __syncthreads();

        // ---- scores = Q @ K^T * 1/sqrt(A); only the two diagonal 8x8 blocks used ----
        {
            v8f acc = {};
            #pragma unroll
            for (int kc = 0; kc < 4; ++kc) {
                const __bf16* pa = &sQw[m * HSTR + kc * 32 + khA];
                v8bf lo = *(const v8bf*)pa;
                v8bf hi = *(const v8bf*)(pa + 16);
                v16bf aq = __builtin_shufflevector(lo, hi, 0, 1, 2, 3, 4, 5, 6, 7,
                                                           8, 9, 10, 11, 12, 13, 14, 15);
                int kk = kc * 32 + kqB;
                v16bf bkt = ld_bf16x16(&sKw[nB * HSTR + kk]);  // row nB of k = column of K^T
                acc = __builtin_amdgcn_wmma_f32_16x16x32_bf16(
                          false, aq, false, bkt, (short)0, acc, false, false);
            }
            const float scale = 0.08838834764831845f;   // 1/sqrt(128)
            #pragma unroll
            for (int r = 0; r < 8; ++r) sSw[(r + mD) * 16 + nB] = acc[r] * scale;
        }

        // ---- v = x @ Wv + bv, one row per lane (16 rows), vectorized ----
        if (lane < 16) {
            const float4* xr  = (const float4*)&sXw[lane * XSTR];
            const float4* wv4 = (const float4*)sWv;
            float4 a4 = {0.f, 0.f, 0.f, 0.f};
            #pragma unroll 8
            for (int f4 = 0; f4 < 32; ++f4) {
                float4 xa = xr[f4], wb = wv4[f4];
                a4.x += xa.x * wb.x; a4.y += xa.y * wb.y;
                a4.z += xa.z * wb.z; a4.w += xa.w * wb.w;
            }
            sVw[lane] = sBv + a4.x + a4.y + a4.z + a4.w;
        }
        __syncthreads();

        // ---- per row: softmax over channel scores, dot with v ----
        if (lane < 16) {
            const int p = lane >> 3;   // token within pair
            float s[8];
            #pragma unroll
            for (int d = 0; d < 8; ++d) s[d] = sSw[lane * 16 + p * 8 + d];
            float mx = s[0];
            #pragma unroll
            for (int d = 1; d < 8; ++d) mx = fmaxf(mx, s[d]);
            float den = 0.f, num = 0.f;
            #pragma unroll
            for (int d = 0; d < 8; ++d) {
                float e = __expf(s[d] - mx);
                den += e;
                num += e * sVw[p * 8 + d];
            }
            sWw[lane] = num / den;     // (softmax(scores) @ v)[row]
        }
        __syncthreads();

        // ---- softmax over channels, weighted channel average, store out ----
        {
            const int p = lane >> 4;   // token within pair
            float w8[8];
            #pragma unroll
            for (int c = 0; c < 8; ++c) w8[c] = sWw[p * 8 + c];
            float mx = w8[0];
            #pragma unroll
            for (int c = 1; c < 8; ++c) mx = fmaxf(mx, w8[c]);
            float den = 0.f;
            #pragma unroll
            for (int c = 0; c < 8; ++c) { w8[c] = __expf(w8[c] - mx); den += w8[c]; }
            const float inv = 1.0f / den;

            float o[8];
            #pragma unroll
            for (int j = 0; j < 8; ++j) o[j] = 0.f;
            const int fbase = (lane & 15) * 8;
            #pragma unroll
            for (int c = 0; c < 8; ++c) {
                const float4* xr = (const float4*)&sXw[(p * 8 + c) * XSTR + fbase];
                float4 x0 = xr[0], x1 = xr[1];
                float wc = w8[c];
                o[0] += wc * x0.x; o[1] += wc * x0.y; o[2] += wc * x0.z; o[3] += wc * x0.w;
                o[4] += wc * x1.x; o[5] += wc * x1.y; o[6] += wc * x1.z; o[7] += wc * x1.w;
            }
            float* og = out + (size_t)(pair * 2 + p) * Fn + fbase;
            float4 o0 = {o[0] * inv, o[1] * inv, o[2] * inv, o[3] * inv};
            float4 o1 = {o[4] * inv, o[5] * inv, o[6] * inv, o[7] * inv};
            *(float4*)&og[0] = o0;
            *(float4*)&og[4] = o1;
        }
        __syncthreads();   // before next iteration overwrites the tiles
    }
}

extern "C" void kernel_launch(void* const* d_in, const int* in_sizes, int n_in,
                              void* d_out, int out_size, void* d_ws, size_t ws_size,
                              hipStream_t stream) {
    (void)in_sizes; (void)n_in; (void)out_size; (void)d_ws; (void)ws_size;
    const float* x  = (const float*)d_in[0];
    const float* Wq = (const float*)d_in[1];
    const float* bq = (const float*)d_in[2];
    const float* Wk = (const float*)d_in[3];
    const float* bk = (const float*)d_in[4];
    const float* Wv = (const float*)d_in[5];
    const float* bv = (const float*)d_in[6];
    float* out = (float*)d_out;
    qkv_chan_attn_kernel<<<BLOCKS, THREADS, 0, stream>>>(x, Wq, bq, Wk, bk, Wv, bv, out);
}
